// VerticalMoE_Predict_Sams_78941498900785
// MI455X (gfx1250) — matrix operand order repaired and checked by reading, not compiled, not run on hardware
//
#include <hip/hip_runtime.h>

#define B_TOT      65536
#define NFIELD     20
#define SQL_NEMB   10
#define DATA_NEMB  10
#define HID        64
#define KEXP       16
#define EXP_IN     200
#define EXP_IN_PAD 224
#define EPSB       1e-5f

typedef __attribute__((ext_vector_type(16))) __bf16        v16bf;
typedef __attribute__((ext_vector_type(8)))  float         v8f;
typedef __attribute__((ext_vector_type(4)))  unsigned int  v4u;

union FragU { v16bf v; v4u u[2]; };

static __device__ __forceinline__ unsigned short f2bf(float f) {
    unsigned int u = __float_as_uint(f);
    u += 0x7FFFu + ((u >> 16) & 1u);           // round-to-nearest-even
    return (unsigned short)(u >> 16);
}

// ---------------------------------------------------------------------------
// Kernel 1: weight transpose -> bf16 column-major copies + zero importance.
// ew1T[k][h][i] (i padded to 224), ew2T[k][g][h], pw1T[g][h]
// ---------------------------------------------------------------------------
__global__ void k_prep(const float* __restrict__ e_w1, const float* __restrict__ e_w2,
                       const float* __restrict__ p_w1,
                       unsigned short* __restrict__ ew1T, unsigned short* __restrict__ ew2T,
                       unsigned short* __restrict__ pw1T, float* __restrict__ imp) {
    int t = blockIdx.x * 256 + threadIdx.x;
    const int N1 = KEXP * HID * EXP_IN_PAD;   // 229376
    const int N2 = KEXP * HID * HID;          // 65536
    const int N3 = HID * HID;                 // 4096
    if (t < N1) {
        int k = t / (HID * EXP_IN_PAD);
        int r = t - k * (HID * EXP_IN_PAD);
        int h = r / EXP_IN_PAD;
        int i = r - h * EXP_IN_PAD;
        float v = (i < EXP_IN) ? e_w1[((size_t)k * EXP_IN + i) * HID + h] : 0.f;
        ew1T[t] = f2bf(v);
    } else if (t < N1 + N2) {
        int u = t - N1;
        int k = u / (HID * HID);
        int r = u - k * (HID * HID);
        int g = r / HID;
        int h = r - g * HID;
        ew2T[u] = f2bf(e_w2[((size_t)k * HID + h) * HID + g]);
    } else if (t < N1 + N2 + N3) {
        int u = t - N1 - N2;
        int g = u / HID;
        int h = u - g * HID;
        pw1T[u] = f2bf(p_w1[h * HID + g]);
    } else if (t < N1 + N2 + N3 + KEXP) {
        imp[t - N1 - N2 - N3] = 0.f;
    }
}

// ---------------------------------------------------------------------------
// Kernel 2: gather x embeddings -> bf16 [B][224] (K zero-padded)
// ---------------------------------------------------------------------------
__global__ void k_gather(const int* __restrict__ x, const float* __restrict__ tab,
                         unsigned short* __restrict__ xemb) {
    int t = blockIdx.x * 256 + threadIdx.x;
    int b = t / EXP_IN_PAD;
    int i = t - b * EXP_IN_PAD;
    float v = 0.f;
    if (i < EXP_IN) {
        int f = i / DATA_NEMB;
        int d = i - f * DATA_NEMB;
        int idx = x[b * NFIELD + f];
        v = tab[(size_t)idx * DATA_NEMB + d];
    }
    xemb[t] = f2bf(v);
}

// ---------------------------------------------------------------------------
// Kernel 3: gate MLP + softmax + top-2 -> dense renormalized gates [B][16];
//           importance accumulation via LDS + global float atomics.
// ---------------------------------------------------------------------------
__global__ __launch_bounds__(256) void k_gate(
        const int* __restrict__ sql, const float* __restrict__ sql_tab,
        const float* __restrict__ gw1, const float* __restrict__ gb1,
        const float* __restrict__ gw2, const float* __restrict__ gb2,
        float* __restrict__ gates, float* __restrict__ imp) {
    __shared__ float simp[KEXP];
    if (threadIdx.x < KEXP) simp[threadIdx.x] = 0.f;
    __syncthreads();

    int b = blockIdx.x * 256 + threadIdx.x;
    float gh[HID];
#pragma unroll
    for (int h = 0; h < HID; ++h) gh[h] = gb1[h];
    for (int f = 0; f < NFIELD; ++f) {
        int idx = sql[b * NFIELD + f];
        const float* e = sql_tab + (size_t)idx * SQL_NEMB;
#pragma unroll
        for (int d = 0; d < SQL_NEMB; ++d) {
            float s = e[d];
            const float* wrow = gw1 + (f * SQL_NEMB + d) * HID;
#pragma unroll
            for (int h = 0; h < HID; ++h) gh[h] = fmaf(s, wrow[h], gh[h]);
        }
    }
#pragma unroll
    for (int h = 0; h < HID; ++h) gh[h] = fmaxf(gh[h], 0.f);

    float sc[KEXP];
#pragma unroll
    for (int k = 0; k < KEXP; ++k) sc[k] = gb2[k];
#pragma unroll
    for (int h = 0; h < HID; ++h) {
        float v = gh[h];
#pragma unroll
        for (int k = 0; k < KEXP; ++k) sc[k] = fmaf(v, gw2[h * KEXP + k], sc[k]);
    }
    float mx = sc[0];
#pragma unroll
    for (int k = 1; k < KEXP; ++k) mx = fmaxf(mx, sc[k]);
    float den = 0.f;
#pragma unroll
    for (int k = 0; k < KEXP; ++k) { sc[k] = __expf(sc[k] - mx); den += sc[k]; }
    float inv = 1.f / den;
#pragma unroll
    for (int k = 0; k < KEXP; ++k) sc[k] *= inv;

    int i1 = 0; float b1 = sc[0];
#pragma unroll
    for (int k = 1; k < KEXP; ++k) if (sc[k] > b1) { b1 = sc[k]; i1 = k; }
    int i2 = -1; float b2 = -1.f;
#pragma unroll
    for (int k = 0; k < KEXP; ++k) if (k != i1 && sc[k] > b2) { b2 = sc[k]; i2 = k; }
    float rs  = 1.f / (b1 + b2 + 1e-9f);
    float g1v = b1 * rs, g2v = b2 * rs;
#pragma unroll
    for (int k = 0; k < KEXP; ++k) {
        float g = (k == i1) ? g1v : ((k == i2) ? g2v : 0.f);
        gates[(size_t)b * KEXP + k] = g;
    }
    atomicAdd(&simp[i1], g1v);
    atomicAdd(&simp[i2], g2v);
    __syncthreads();
    if (threadIdx.x < KEXP) atomicAdd(&imp[threadIdx.x], simp[threadIdx.x]);
}

// ---------------------------------------------------------------------------
// Kernel 4: MoE experts + combine + predict head. One wave per 16-row tile.
// ---------------------------------------------------------------------------
__global__ __launch_bounds__(256) void k_moe(
        const unsigned short* __restrict__ xemb, const float* __restrict__ gates,
        const unsigned short* __restrict__ ew1T, const unsigned short* __restrict__ ew2T,
        const unsigned short* __restrict__ pw1T,
        const float* __restrict__ e_b1, const float* __restrict__ e_g1, const float* __restrict__ e_be1,
        const float* __restrict__ e_m1, const float* __restrict__ e_v1,
        const float* __restrict__ e_b2, const float* __restrict__ e_g2, const float* __restrict__ e_be2,
        const float* __restrict__ e_m2, const float* __restrict__ e_v2,
        const float* __restrict__ p_b1, const float* __restrict__ p_gv, const float* __restrict__ p_be,
        const float* __restrict__ p_m,  const float* __restrict__ p_v,
        const float* __restrict__ p_w2, const float* __restrict__ p_b2,
        float* __restrict__ out) {
    __shared__ __align__(16) unsigned short shH[8][16 * 72];   // per-wave 16x64 bf16 (stride 72)
    __shared__ float shG[8][16 * KEXP];                        // per-wave gate tile

    const int lane = threadIdx.x & 31;
    const int w    = threadIdx.x >> 5;
    const int tile = blockIdx.x * 8 + w;
    const int colL = lane & 15;
    const int hb   = lane >> 4;          // which half-wave
    const int aoff = hb * 8;             // A frag: u0 @ {0|8}, u1 @ +16  (ISA 7.12.2)
    const int boff = hb * 8;             // B frag: u0 @ {0|8}, u1 @ +16 (same K order as A)

    // stage this tile's gates into LDS
    {
        const float* gsrc = gates + (size_t)tile * 16 * KEXP;
#pragma unroll
        for (int j = 0; j < 8; ++j) shG[w][lane * 8 + j] = gsrc[lane * 8 + j];
    }
    asm volatile("s_wait_dscnt 0" ::: "memory");

    const v8f vzero = {0.f, 0.f, 0.f, 0.f, 0.f, 0.f, 0.f, 0.f};
    v8f y[4] = {vzero, vzero, vzero, vzero};

    const unsigned short* arow = xemb + ((size_t)tile * 16 + colL) * EXP_IN_PAD;

    for (int k = 0; k < KEXP; ++k) {
        // wave-uniform expert skip: any of the 16 rows routed to expert k?
        bool mine = (hb == 0) && (shG[w][colL * KEXP + k] > 0.f);
        if (__ballot(mine) == 0ull) continue;

        // ---- GEMM1: h1 = x_emb(16x224) x e_w1[k](224x64), 7 k-steps x 4 n-tiles
        v8f acc[4] = {vzero, vzero, vzero, vzero};
        for (int kt = 0; kt < 7; ++kt) {
            FragU a;
            const unsigned short* ap = arow + kt * 32 + aoff;
            a.u[0] = *(const v4u*)(ap);
            a.u[1] = *(const v4u*)(ap + 16);
#pragma unroll
            for (int n = 0; n < 4; ++n) {
                FragU bf;
                const unsigned short* bp =
                    ew1T + ((size_t)(k * HID + n * 16 + colL)) * EXP_IN_PAD + kt * 32 + boff;
                bf.u[0] = *(const v4u*)(bp);
                bf.u[1] = *(const v4u*)(bp + 16);
                acc[n] = __builtin_amdgcn_wmma_f32_16x16x32_bf16(
                    false, a.v, false, bf.v, (short)0, acc[n], false, false);
            }
        }
        // BN1 + ReLU -> restage as bf16 A-operand in LDS
#pragma unroll
        for (int n = 0; n < 4; ++n) {
            int col = n * 16 + colL;
            int pi  = k * HID + col;
            float scale = e_g1[pi] * rsqrtf(e_v1[pi] + EPSB);
            float shift = (e_b1[pi] - e_m1[pi]) * scale + e_be1[pi];
#pragma unroll
            for (int r = 0; r < 8; ++r) {
                float val = fmaxf(fmaf(acc[n][r], scale, shift), 0.f);
                shH[w][(r + hb * 8) * 72 + col] = f2bf(val);
            }
        }
        asm volatile("s_wait_dscnt 0" ::: "memory");

        // ---- GEMM2: h2 = h1(16x64) x e_w2[k](64x64)
        v8f acc2[4] = {vzero, vzero, vzero, vzero};
#pragma unroll
        for (int kt = 0; kt < 2; ++kt) {
            FragU a;
            const unsigned short* ap = &shH[w][colL * 72 + kt * 32 + aoff];
            a.u[0] = *(const v4u*)(ap);
            a.u[1] = *(const v4u*)(ap + 16);
#pragma unroll
            for (int n = 0; n < 4; ++n) {
                FragU bf;
                const unsigned short* bp =
                    ew2T + ((size_t)(k * HID + n * 16 + colL)) * HID + kt * 32 + boff;
                bf.u[0] = *(const v4u*)(bp);
                bf.u[1] = *(const v4u*)(bp + 16);
                acc2[n] = __builtin_amdgcn_wmma_f32_16x16x32_bf16(
                    false, a.v, false, bf.v, (short)0, acc2[n], false, false);
            }
        }
        // BN2 + ReLU + gate-weighted accumulate into y
        float gw[8];
#pragma unroll
        for (int r = 0; r < 8; ++r) gw[r] = shG[w][(r + hb * 8) * KEXP + k];
#pragma unroll
        for (int n = 0; n < 4; ++n) {
            int col = n * 16 + colL;
            int pi  = k * HID + col;
            float scale = e_g2[pi] * rsqrtf(e_v2[pi] + EPSB);
            float shift = (e_b2[pi] - e_m2[pi]) * scale + e_be2[pi];
#pragma unroll
            for (int r = 0; r < 8; ++r) {
                float val = fmaxf(fmaf(acc2[n][r], scale, shift), 0.f);
                y[n][r] = fmaf(gw[r], val, y[n][r]);
            }
        }
    }

    // ---- predict head: ph = relu(bn(y @ p_w1)); out = ph @ p_w2 + p_b2
#pragma unroll
    for (int n = 0; n < 4; ++n) {
        int col = n * 16 + colL;
#pragma unroll
        for (int r = 0; r < 8; ++r)
            shH[w][(r + hb * 8) * 72 + col] = f2bf(y[n][r]);
    }
    asm volatile("s_wait_dscnt 0" ::: "memory");

    v8f acc3[4] = {vzero, vzero, vzero, vzero};
#pragma unroll
    for (int kt = 0; kt < 2; ++kt) {
        FragU a;
        const unsigned short* ap = &shH[w][colL * 72 + kt * 32 + aoff];
        a.u[0] = *(const v4u*)(ap);
        a.u[1] = *(const v4u*)(ap + 16);
#pragma unroll
        for (int n = 0; n < 4; ++n) {
            FragU bf;
            const unsigned short* bp = pw1T + (size_t)(n * 16 + colL) * HID + kt * 32 + boff;
            bf.u[0] = *(const v4u*)(bp);
            bf.u[1] = *(const v4u*)(bp + 16);
            acc3[n] = __builtin_amdgcn_wmma_f32_16x16x32_bf16(
                false, a.v, false, bf.v, (short)0, acc3[n], false, false);
        }
    }
    float part[8] = {0.f, 0.f, 0.f, 0.f, 0.f, 0.f, 0.f, 0.f};
#pragma unroll
    for (int n = 0; n < 4; ++n) {
        int col = n * 16 + colL;
        float scale = p_gv[col] * rsqrtf(p_v[col] + EPSB);
        float shift = (p_b1[col] - p_m[col]) * scale + p_be[col];
        float w2v   = p_w2[col];
#pragma unroll
        for (int r = 0; r < 8; ++r) {
            float val = fmaxf(fmaf(acc3[n][r], scale, shift), 0.f);
            part[r] = fmaf(val, w2v, part[r]);
        }
    }
#pragma unroll
    for (int m = 1; m < 16; m <<= 1) {
#pragma unroll
        for (int r = 0; r < 8; ++r) part[r] += __shfl_xor(part[r], m, 32);
    }
    if (colL == 0) {
        float bias = p_b2[0];
        int rb = tile * 16 + hb * 8;
#pragma unroll
        for (int r = 0; r < 8; ++r) out[rb + r] = part[r] + bias;
    }
}

// ---------------------------------------------------------------------------
// Kernel 5: load-balance loss = var(importance) / (mean^2 + 1e-10)
// ---------------------------------------------------------------------------
__global__ void k_loss(const float* __restrict__ imp, float* __restrict__ loss_out) {
    if (threadIdx.x == 0) {
        float s = 0.f;
        for (int k = 0; k < KEXP; ++k) s += imp[k];
        float mean = s / (float)KEXP;
        float var = 0.f;
        for (int k = 0; k < KEXP; ++k) { float d = imp[k] - mean; var += d * d; }
        var /= (float)KEXP;
        loss_out[0] = var / (mean * mean + 1e-10f);
    }
}

extern "C" void kernel_launch(void* const* d_in, const int* in_sizes, int n_in,
                              void* d_out, int out_size, void* d_ws, size_t ws_size,
                              hipStream_t stream) {
    (void)in_sizes; (void)n_in; (void)out_size; (void)ws_size;
    const int*   x        = (const int*)d_in[0];
    const int*   sql      = (const int*)d_in[1];
    const float* in_tab   = (const float*)d_in[2];
    const float* sql_tab  = (const float*)d_in[3];
    const float* gate_w1  = (const float*)d_in[4];
    const float* gate_b1  = (const float*)d_in[5];
    const float* gate_w2  = (const float*)d_in[6];
    const float* gate_b2  = (const float*)d_in[7];
    const float* e_w1     = (const float*)d_in[8];
    const float* e_b1     = (const float*)d_in[9];
    const float* e_g1     = (const float*)d_in[10];
    const float* e_be1    = (const float*)d_in[11];
    const float* e_m1     = (const float*)d_in[12];
    const float* e_v1     = (const float*)d_in[13];
    const float* e_w2     = (const float*)d_in[14];
    const float* e_b2     = (const float*)d_in[15];
    const float* e_g2     = (const float*)d_in[16];
    const float* e_be2    = (const float*)d_in[17];
    const float* e_m2     = (const float*)d_in[18];
    const float* e_v2     = (const float*)d_in[19];
    const float* p_w1     = (const float*)d_in[20];
    const float* p_b1     = (const float*)d_in[21];
    const float* p_g      = (const float*)d_in[22];
    const float* p_be     = (const float*)d_in[23];
    const float* p_m      = (const float*)d_in[24];
    const float* p_v      = (const float*)d_in[25];
    const float* p_w2     = (const float*)d_in[26];
    const float* p_b2     = (const float*)d_in[27];
    float* out = (float*)d_out;

    char* ws = (char*)d_ws;
    unsigned short* xemb  = (unsigned short*)(ws);               // B*224*2   = 29,360,128
    float*          gatesW = (float*)(ws + 29360128);            // B*16*4    =  4,194,304
    float*          imp    = (float*)(ws + 33554432);            // 256-byte slot
    unsigned short* ew1T   = (unsigned short*)(ws + 33554688);   // 16*64*224*2 = 458,752
    unsigned short* ew2T   = (unsigned short*)(ws + 34013440);   // 16*64*64*2  = 131,072
    unsigned short* pw1T   = (unsigned short*)(ws + 34144512);   // 64*64*2     =   8,192

    k_prep<<<(299024 + 255) / 256, 256, 0, stream>>>(e_w1, e_w2, p_w1, ew1T, ew2T, pw1T, imp);
    k_gather<<<(B_TOT * EXP_IN_PAD) / 256, 256, 0, stream>>>(x, in_tab, xemb);
    k_gate<<<B_TOT / 256, 256, 0, stream>>>(sql, sql_tab, gate_w1, gate_b1, gate_w2, gate_b2,
                                            gatesW, imp);
    k_moe<<<(B_TOT / 16) / 8, 256, 0, stream>>>(xemb, gatesW, ew1T, ew2T, pw1T,
        e_b1, e_g1, e_be1, e_m1, e_v1,
        e_b2, e_g2, e_be2, e_m2, e_v2,
        p_b1, p_g, p_be, p_m, p_v, p_w2, p_b2, out);
    k_loss<<<1, 32, 0, stream>>>(imp, out + B_TOT);
}
